// VKDE_26680336843081
// MI455X (gfx1250) — matrix operand order repaired and checked by reading, not compiled, not run on hardware
//
#include <hip/hip_runtime.h>
#include <cstdint>
#include <cstddef>

// ---- problem constants (match reference) ----
#define cN 20000   // NUM_ITEMS
#define cK 100     // TOPK
#define cB 128     // B
#define cT 4096    // T
#define cH 600     // H
#define cD 200     // D
#define KP 224     // K padded to 7*32 for wmma f16 k-steps
#define LDSP 232   // padded LDS row stride (halves): 464B -> conflict-free 16B reads
#define NPAD 20032 // 313 * 64 item rows (padded, zero-filled tail)
#define EPSV 1e-12f
#define INV_TAU 5.0f

typedef _Float16 v8h  __attribute__((ext_vector_type(8)));
typedef _Float16 v16h __attribute__((ext_vector_type(16)));
typedef float    v8f  __attribute__((ext_vector_type(8)));

// ---------------- zero fill ----------------
__global__ void k_zero(float* __restrict__ p, size_t n) {
    size_t i = (size_t)blockIdx.x * blockDim.x + threadIdx.x;
    size_t stride = (size_t)gridDim.x * blockDim.x;
    for (; i < n; i += stride) p[i] = 0.0f;
}

// ---------------- sparse row preprocess: mask, dedup, L1, L2, scatter x ----------------
__global__ __launch_bounds__(128) void k_sparse(
    const float* __restrict__ rating, const int* __restrict__ gram_idx,
    const float* __restrict__ gram_val, const int* __restrict__ item_idx,
    const int* __restrict__ segids,
    int* __restrict__ cidx, float* __restrict__ cw,
    float* __restrict__ xout, float* __restrict__ counts) {
    const int t = blockIdx.x;
    const int k = threadIdx.x;
    __shared__ int   sidx[cK];
    __shared__ float sc[cK];
    __shared__ float red[128];

    const int it = item_idx[t];
    const int sb = segids[t];

    float c = 0.0f; int id = 0;
    if (k < cK) {
        id = gram_idx[(size_t)it * cK + k];
        float val = gram_val[(size_t)it * cK + k];
        float r = rating[(size_t)sb * cN + id];
        c = (r > 0.0f) ? val : 0.0f;     // mask = (l2norm(rating) > 0) <=> rating > 0
        sidx[k] = id; sc[k] = c;
    }
    __syncthreads();

    // L1 denom: all masked values >= 0, so sum|.| == plain sum (duplicates OK)
    red[k] = c; __syncthreads();
    for (int s = 64; s > 0; s >>= 1) { if (k < s) red[k] += red[k + s]; __syncthreads(); }
    const float S1 = red[0];
    __syncthreads();

    // dedup: first occurrence owns the combined value (needed for exact L2)
    bool keeper = false; float comb = 0.0f;
    if (k < cK) {
        int first = k;
        for (int j = 0; j < k; ++j) if (sidx[j] == id) { first = j; break; }
        if (first == k) {
            keeper = true;
            for (int j = k; j < cK; ++j) if (sidx[j] == id) comb += sc[j];
        }
    }
    const float l1 = comb / fmaxf(S1, EPSV);
    __syncthreads();
    red[k] = l1 * l1; __syncthreads();
    for (int s = 64; s > 0; s >>= 1) { if (k < s) red[k] += red[k + s]; __syncthreads(); }
    const float S2 = sqrtf(red[0]);
    const float wv = l1 / fmaxf(S2, EPSV);

    if (k < cK) {
        cidx[(size_t)t * cK + k] = id;
        cw[(size_t)t * cK + k]   = keeper ? wv : 0.0f;
        if (keeper) xout[(size_t)t * cN + id] = wv;   // x buffer pre-zeroed
    }
    if (k == 0) atomicAdd(&counts[sb], 1.0f);
}

// ---------------- h = tanh(x @ W1 + b1), sparse gather (W1 is L2-resident) ----------------
__global__ __launch_bounds__(256) void k_h(
    const int* __restrict__ cidx, const float* __restrict__ cw,
    const float* __restrict__ W1, const float* __restrict__ b1,
    float* __restrict__ h) {
    const int t = blockIdx.x;
    const int tid = threadIdx.x;
    __shared__ int   si[cK];
    __shared__ float sw[cK];
    if (tid < cK) { si[tid] = cidx[(size_t)t * cK + tid]; sw[tid] = cw[(size_t)t * cK + tid]; }
    __syncthreads();
    for (int j = tid; j < cH; j += 256) {
        float acc = b1[j];
        #pragma unroll 4
        for (int k = 0; k < cK; ++k) acc += sw[k] * W1[(size_t)si[k] * cH + j];
        h[(size_t)t * cH + j] = tanhf(acc);
    }
}

// ---------------- normalize items -> f16, K padded, rows padded to NPAD ----------------
__global__ __launch_bounds__(256) void k_itemnorm(
    const float* __restrict__ items, _Float16* __restrict__ itn) {
    const int r = blockIdx.x;
    const int tid = threadIdx.x;
    __shared__ float red[256];
    __shared__ float sinv;
    float v = 0.0f;
    if (r < cN && tid < cD) v = items[(size_t)r * cD + tid];
    red[tid] = v * v; __syncthreads();
    for (int s = 128; s > 0; s >>= 1) { if (tid < s) red[tid] += red[tid + s]; __syncthreads(); }
    if (tid == 0) sinv = 1.0f / fmaxf(sqrtf(red[0]), EPSV);
    __syncthreads();
    if (tid < KP) {
        float o = (r < cN && tid < cD) ? v * sinv : 0.0f;
        itn[(size_t)r * KP + tid] = (_Float16)o;
    }
}

// ---------------- mz = h @ W2 + b2 ; z ; KL partials ; zn (f16, padded) ----------------
__global__ __launch_bounds__(256) void k_mlp2(
    const float* __restrict__ h, const float* __restrict__ W2,
    const float* __restrict__ b2, const float* __restrict__ epsn,
    float* __restrict__ zout, _Float16* __restrict__ zn,
    float* __restrict__ klacc) {
    const int t = blockIdx.x;
    const int tid = threadIdx.x;
    __shared__ float hs[cH];
    __shared__ float mzs[2 * cD];
    __shared__ float red[256];
    __shared__ float sinv;

    for (int k = tid; k < cH; k += 256) hs[k] = h[(size_t)t * cH + k];
    __syncthreads();
    for (int j = tid; j < 2 * cD; j += 256) {
        float acc = b2[j];
        #pragma unroll 4
        for (int k = 0; k < cH; ++k) acc += hs[k] * W2[(size_t)k * (2 * cD) + j];
        mzs[j] = acc;
    }
    __syncthreads();

    float zv = 0.0f, kt = 0.0f;
    if (tid < cD) {
        float mean = mzs[tid], lv = mzs[cD + tid];
        float ev = __expf(lv);
        zv = mean + epsn[(size_t)t * cD + tid] * __expf(0.5f * lv);
        zout[(size_t)t * cD + tid] = zv;
        kt = mean * mean + ev - 1.0f - lv;
    }
    red[tid] = kt; __syncthreads();
    for (int s = 128; s > 0; s >>= 1) { if (tid < s) red[tid] += red[tid + s]; __syncthreads(); }
    if (tid == 0) atomicAdd(klacc, red[0]);
    __syncthreads();
    red[tid] = zv * zv; __syncthreads();
    for (int s = 128; s > 0; s >>= 1) { if (tid < s) red[tid] += red[tid + s]; __syncthreads(); }
    if (tid == 0) sinv = 1.0f / fmaxf(sqrtf(red[0]), EPSV);
    __syncthreads();
    if (tid < KP) zn[(size_t)t * KP + tid] = (_Float16)((tid < cD) ? zv * sinv : 0.0f);
}

// ---------------- big similarity GEMM via WMMA f16 + exp + segment-sum ----------------
// grid = (313, 128); block = 256 (8 wave32). Block tile: 32 t x 64 i.
// A (32x224) and B (64x224) f16 tiles are staged into LDS with
// global_load_async_to_lds_b128 (ASYNCcnt path), then each wave runs a
// 16x16 WMMA tile with K loop 7 x 32 reading fragments from LDS.
__global__ __launch_bounds__(256) void k_sim(
    const _Float16* __restrict__ zn, const _Float16* __restrict__ itn,
    const int* __restrict__ segids, float* __restrict__ segbuf) {
    __shared__ _Float16 As[32][LDSP];   // 14.8 KB
    __shared__ _Float16 Bs[64][LDSP];   // 29.7 KB
    __shared__ int sld[32];

    const int t0b = blockIdx.y * 32;
    const int i0b = blockIdx.x * 64;
    if (threadIdx.x < 32) sld[threadIdx.x] = segids[t0b + threadIdx.x];

    // ---- async-stage A tile: 32 rows x 28 16B-chunks ----
    for (int u = threadIdx.x; u < 32 * 28; u += 256) {
        const int row = u / 28, ch = u % 28;
        unsigned ldst = (unsigned)(size_t)(&As[row][ch * 8]);
        unsigned long long ga =
            (unsigned long long)(size_t)(zn + (size_t)(t0b + row) * KP + ch * 8);
        asm volatile("global_load_async_to_lds_b128 %0, %1, off"
                     :: "v"(ldst), "v"(ga) : "memory");
    }
    // ---- async-stage B tile: 64 rows x 28 16B-chunks ----
    for (int u = threadIdx.x; u < 64 * 28; u += 256) {
        const int row = u / 28, ch = u % 28;
        unsigned ldst = (unsigned)(size_t)(&Bs[row][ch * 8]);
        unsigned long long ga =
            (unsigned long long)(size_t)(itn + (size_t)(i0b + row) * KP + ch * 8);
        asm volatile("global_load_async_to_lds_b128 %0, %1, off"
                     :: "v"(ldst), "v"(ga) : "memory");
    }
    asm volatile("s_wait_asynccnt 0" ::: "memory");
    __syncthreads();

    const int w    = threadIdx.x >> 5;
    const int lane = threadIdx.x & 31;
    const int trow = (w >> 2) * 16;                 // 0 or 16 within block tile
    const int hl   = lane >> 4;                     // lane half (0/1)
    const int ln   = lane & 15;

    // A frag (16x32 f16): lane holds row trow+ln; K = {0..7,16..23} (+8 for hi half)
    const _Float16* arow = &As[trow + ln][hl * 8];
    // B frag (32x16 f16): lane holds column; K = {0..7,16..23} (+8 for hi half)
    const _Float16* brow = &Bs[(w & 3) * 16 + ln][hl * 8];

    v8f acc = {};
    for (int kk = 0; kk < KP; kk += 32) {
        v8h a0 = *(const v8h*)(arow + kk);
        v8h a1 = *(const v8h*)(arow + kk + 16);
        v16h a = __builtin_shufflevector(a0, a1, 0,1,2,3,4,5,6,7,8,9,10,11,12,13,14,15);
        v8h b0 = *(const v8h*)(brow + kk);
        v8h b1 = *(const v8h*)(brow + kk + 16);
        v16h b = __builtin_shufflevector(b0, b1, 0,1,2,3,4,5,6,7,8,9,10,11,12,13,14,15);
        acc = __builtin_amdgcn_wmma_f32_16x16x32_f16(
            /*neg_a=*/false, a, /*neg_b=*/false, b,
            /*c_mod=*/(short)0, acc, /*reuse_a=*/false, /*reuse_b=*/false);
    }

    const int i = i0b + (w & 3) * 16 + ln;
    if (i < cN) {
        // C layout: vgpr v, lane half hl -> row M = v + 8*hl (8 consecutive t's per lane)
        const int rbase = trow + hl * 8;
        int cur = sld[rbase];
        float run = 0.0f;
        #pragma unroll
        for (int v = 0; v < 8; ++v) {
            int s = sld[rbase + v];
            float e = __expf(acc[v] * INV_TAU);
            if (s != cur) {
                atomicAdd(&segbuf[(size_t)cur * cN + i], run);
                cur = s; run = 0.0f;
            }
            run += e;
        }
        atomicAdd(&segbuf[(size_t)cur * cN + i], run);
    }
}

// ---------------- finalize: new_output and kl ----------------
__global__ void k_final(const float* __restrict__ segbuf, const float* __restrict__ counts,
                        const float* __restrict__ klacc,
                        float* __restrict__ newout, float* __restrict__ klout) {
    size_t id = (size_t)blockIdx.x * blockDim.x + threadIdx.x;
    if (id == 0) klout[0] = 0.5f * klacc[0] / (float)cT;
    if (id < (size_t)cB * cN) {
        int b = (int)(id / cN);
        float cnt = counts[b];
        float v = 0.0f;
        if (cnt > 0.0f) v = logf(segbuf[id] / fmaxf(cnt, 1.0f) + 1.0f);
        newout[id] = v;
    }
}

extern "C" void kernel_launch(void* const* d_in, const int* in_sizes, int n_in,
                              void* d_out, int out_size, void* d_ws, size_t ws_size,
                              hipStream_t stream) {
    (void)in_sizes; (void)n_in; (void)out_size; (void)ws_size;
    const float* rating   = (const float*)d_in[0];
    const int*   gram_idx = (const int*)  d_in[1];
    const float* gram_val = (const float*)d_in[2];
    const int*   item_idx = (const int*)  d_in[3];
    const int*   segids   = (const int*)  d_in[4];
    const float* W1       = (const float*)d_in[5];
    const float* b1       = (const float*)d_in[6];
    const float* W2       = (const float*)d_in[7];
    const float* b2       = (const float*)d_in[8];
    const float* items    = (const float*)d_in[9];
    const float* epsn     = (const float*)d_in[10];

    // output partition (tuple flat, return order: z, new_output, kl, x)
    float* zout   = (float*)d_out;
    float* newout = zout + (size_t)cT * cD;
    float* klout  = newout + (size_t)cB * cN;
    float* xout   = klout + 1;

    // workspace carve (256B aligned)
    char* ws = (char*)d_ws;
    size_t off = 0;
    auto carve = [&](size_t bytes) { size_t o = off; off = (off + bytes + 255) & ~(size_t)255; return o; };
    int*      cidx  = (int*)     (ws + carve((size_t)cT * cK * sizeof(int)));
    float*    cw    = (float*)   (ws + carve((size_t)cT * cK * sizeof(float)));
    float*    hbuf  = (float*)   (ws + carve((size_t)cT * cH * sizeof(float)));
    _Float16* zn    = (_Float16*)(ws + carve((size_t)cT * KP * sizeof(_Float16)));
    _Float16* itn   = (_Float16*)(ws + carve((size_t)NPAD * KP * sizeof(_Float16)));
    float*    segbf = (float*)   (ws + carve((size_t)cB * cN * sizeof(float)));
    float*    cnts  = (float*)   (ws + carve((size_t)(cB + 1) * sizeof(float)));
    float*    klacc = cnts + cB;

    // zero-fill: x output, segment accumulators, counts+klacc
    k_zero<<<2048, 256, 0, stream>>>(xout,  (size_t)cT * cN);
    k_zero<<<2048, 256, 0, stream>>>(segbf, (size_t)cB * cN);
    k_zero<<<1,    256, 0, stream>>>(cnts,  (size_t)(cB + 1));

    k_sparse<<<cT, 128, 0, stream>>>(rating, gram_idx, gram_val, item_idx, segids,
                                     cidx, cw, xout, cnts);
    k_h<<<cT, 256, 0, stream>>>(cidx, cw, W1, b1, hbuf);
    k_itemnorm<<<NPAD, 256, 0, stream>>>(items, itn);
    k_mlp2<<<cT, 256, 0, stream>>>(hbuf, W2, b2, epsn, zout, zn, klacc);

    dim3 gsim(313, 128);
    k_sim<<<gsim, 256, 0, stream>>>(zn, itn, segids, segbf);

    k_final<<<((size_t)cB * cN + 255) / 256, 256, 0, stream>>>(segbf, cnts, klacc, newout, klout);
}
